// OptimizedHybridReasoningLayer_28982439314097
// MI455X (gfx1250) — compile-verified, matrix operand order checked by reading, not run on hardware
//
#include <hip/hip_runtime.h>
#include <math.h>

typedef __attribute__((ext_vector_type(16))) _Float16 v16h;
typedef __attribute__((ext_vector_type(8)))  _Float16 v8h;
typedef __attribute__((ext_vector_type(8)))  float    v8f;

constexpr int kB  = 32;
constexpr int kT  = 512;
constexpr int kD  = 256;
constexpr int kH  = 4;
constexpr int kDH = 64;
constexpr int kW  = 8;
constexpr int kBT  = kB * kT;          // 16384
constexpr long kBTD = (long)kBT * kD;  // 4194304
constexpr int kBD  = kB * kD;          // 8192

// ---------------------------------------------------------------- helpers

__device__ inline float geluf(float x) {
  return 0.5f * x * (1.0f + erff(x * 0.70710678118654752f));
}
__device__ inline float fmapf(float x) {  // elu(x)+1
  return x > 0.0f ? x + 1.0f : expf(x);
}

// A or B(weight) fragment for v_wmma_f32_16x16x32_f16 from a row-major
// [rows x ld] f16 matrix where the contraction dim is contiguous.
__device__ inline v16h frag_rowmajor(const _Float16* base, int row0, int ld, int k0) {
  int lane = threadIdx.x & 31;
  int r = lane & 15, hh = lane >> 4;
  const _Float16* p = base + (size_t)(row0 + r) * (size_t)ld + (k0 + 8 * hh);
  union { v16h v; v8h h[2]; } u;
  u.h[0] = *reinterpret_cast<const v8h*>(p);
  u.h[1] = *reinterpret_cast<const v8h*>(p + 16);
  return u.v;
}

// B fragment gathered from a K-major-row matrix: element (k,n) at base[(k0+k)*ld + n0+n]
__device__ inline v16h frag_gather(const _Float16* base, int k0, int ld, int n0) {
  int lane = threadIdx.x & 31;
  int n = lane & 15, hh = lane >> 4;
  union { v16h v; _Float16 e[16]; } u;
#pragma unroll
  for (int i = 0; i < 16; ++i) {
    int k = ((i < 8) ? i : (i + 8)) + 8 * hh;
    u.e[i] = base[(size_t)(k0 + k) * (size_t)ld + n0 + n];
  }
  return u.v;
}

__device__ inline v8f wmma16(v16h a, v16h b, v8f c) {
  return __builtin_amdgcn_wmma_f32_16x16x32_f16(false, a, false, b, (short)0, c, false, false);
}

// One 16x16 tile of C = A @ W^T ; software-pipelined so loads overlap WMMA issue
__device__ inline v8f tile_mm(const _Float16* A, int lda, int m0,
                              const _Float16* W, int ldw, int n0, int K) {
  v8f acc = {};
  v16h a0 = frag_rowmajor(A, m0, lda, 0);
  v16h b0 = frag_rowmajor(W, n0, ldw, 0);
  for (int k = 0; k < K; k += 32) {
    v16h a1 = a0, b1 = b0;
    if (k + 32 < K) {
      a1 = frag_rowmajor(A, m0, lda, k + 32);
      b1 = frag_rowmajor(W, n0, ldw, k + 32);
    }
    acc = wmma16(a0, b0, acc);
    a0 = a1; b0 = b1;
  }
  return acc;
}

// -------------------- Tensor Data Mover (CDNA5 TDM) --------------------
#if __has_builtin(__builtin_amdgcn_tensor_load_to_lds)
#define SCAN_TDM 1
typedef __attribute__((ext_vector_type(4))) unsigned int v4u;
typedef __attribute__((ext_vector_type(4))) int v4i;
typedef __attribute__((ext_vector_type(8))) int v8i;

// 2D tile load Global->LDS: rows x dim0 elements of f16, row stride stride0 (elems)
__device__ inline void tdm_load_tile_f16(unsigned lds_addr, const _Float16* g,
                                         unsigned dim0, unsigned rows,
                                         unsigned long stride0) {
  unsigned long ga = (unsigned long)(size_t)g;
  v4u g0;
  g0[0] = 1u;                                   // count=1, user descriptor
  g0[1] = lds_addr;                             // LDS byte address
  g0[2] = (unsigned)ga;                         // global_addr[31:0]
  g0[3] = (unsigned)((ga >> 32) & 0x1ffffffu) | 0x80000000u;  // ga[56:32] | type=2
  v8i g1;
  g1[0] = 0x00010000;                           // data_size=1 (2 bytes), no mask
  g1[1] = (int)((dim0 & 0xffffu) << 16);        // tensor_dim0[15:0]
  g1[2] = (int)((dim0 >> 16) | ((rows & 0xffffu) << 16));   // dim0 hi | tensor_dim1 lo
  g1[3] = (int)((rows >> 16) | ((dim0 & 0xffffu) << 16));   // dim1 hi | tile_dim0
  g1[4] = (int)(rows & 0xffffu);                // tile_dim1 (tile_dim2 = 0)
  g1[5] = (int)(unsigned)(stride0 & 0xffffffffu);           // dim0_stride[31:0]
  g1[6] = (int)(unsigned)((stride0 >> 32) & 0xffffu);       // dim0_stride[47:32]
  g1[7] = 0;                                    // dim1_stride = 0
  v4i z4 = {};
#if __clang_major__ >= 23
  v8i z8 = {};
  __builtin_amdgcn_tensor_load_to_lds(g0, g1, z4, z4, z8, 0);
#else
  __builtin_amdgcn_tensor_load_to_lds(g0, g1, z4, z4, 0);
#endif
}
#endif

__device__ inline void wait_tensorcnt0() {
#if __has_builtin(__builtin_amdgcn_s_wait_tensorcnt)
  __builtin_amdgcn_s_wait_tensorcnt(0);
#else
  __asm__ volatile("s_wait_tensorcnt 0x0" ::: "memory");
#endif
}
__device__ inline void wait_tensorcnt1() {
#if __has_builtin(__builtin_amdgcn_s_wait_tensorcnt)
  __builtin_amdgcn_s_wait_tensorcnt(1);
#else
  __asm__ volatile("s_wait_tensorcnt 0x1" ::: "memory");
#endif
}

// ---------------------------------------------------------------- kernels

__global__ void cast_f16_kernel(const float* __restrict__ s, _Float16* __restrict__ d, int n) {
  int i = blockIdx.x * blockDim.x + threadIdx.x;
  if (i < n) d[i] = (_Float16)s[i];
}

// C[M,N] = act(A[M,K] @ W[N,K]^T + bias) ; wave computes 16x64, pipelined k-loop
__global__ void gemm_nt_kernel(const _Float16* __restrict__ A, int lda,
                               const _Float16* __restrict__ W, int ldw,
                               const float* __restrict__ bias,
                               float* __restrict__ Cf, _Float16* __restrict__ Ch,
                               int M, int N, int K, int act) {
  int wid = (blockIdx.x * blockDim.x + threadIdx.x) >> 5;
  int nt4 = N >> 6;
  int total = (M >> 4) * nt4;
  if (wid >= total) return;
  int mt = wid / nt4, nt = wid % nt4;
  int m0 = mt << 4, n0 = nt << 6;
  v8f accs[4] = {};
  v16h a0 = frag_rowmajor(A, m0, lda, 0);
  v16h b0[4];
#pragma unroll
  for (int j = 0; j < 4; ++j) b0[j] = frag_rowmajor(W, n0 + j * 16, ldw, 0);
  for (int k = 0; k < K; k += 32) {
    v16h a1 = a0;
    v16h b1[4];
#pragma unroll
    for (int j = 0; j < 4; ++j) b1[j] = b0[j];
    if (k + 32 < K) {
      a1 = frag_rowmajor(A, m0, lda, k + 32);
#pragma unroll
      for (int j = 0; j < 4; ++j) b1[j] = frag_rowmajor(W, n0 + j * 16, ldw, k + 32);
    }
#pragma unroll
    for (int j = 0; j < 4; ++j) accs[j] = wmma16(a0, b0[j], accs[j]);
    a0 = a1;
#pragma unroll
    for (int j = 0; j < 4; ++j) b0[j] = b1[j];
  }
  int lane = threadIdx.x & 31;
  int nn = lane & 15, hh = lane >> 4;
#pragma unroll
  for (int j = 0; j < 4; ++j) {
#pragma unroll
    for (int r = 0; r < 8; ++r) {
      int row = m0 + r + 8 * hh;
      int col = n0 + j * 16 + nn;
      float v = accs[j][r] + (bias ? bias[col] : 0.0f);
      if (act == 1) v = geluf(v);
      if (Cf) Cf[(size_t)row * N + col] = v;
      if (Ch) Ch[(size_t)row * N + col] = (_Float16)v;
    }
  }
}

// flash-style causal attention; one wave per (b,h,16-query tile). qkvh:[B,T,3D] f16.
__global__ void flash_attn_kernel(const _Float16* __restrict__ qkvh,
                                  _Float16* __restrict__ aoh) {
  __shared__ __align__(16) _Float16 pbuf[4][16 * 32];
  int wid = threadIdx.x >> 5;
  int gw = blockIdx.x * 4 + wid;                 // 0 .. B*H*32-1
  int mt = gw & 31;
  int bh = gw >> 5;
  int h = bh & 3, b = bh >> 2;
  const int ld = 3 * kD;
  const _Float16* Qb = qkvh + (size_t)b * kT * ld + h * kDH;
  const _Float16* Kb = Qb + kD;
  const _Float16* Vb = Qb + 2 * kD;
  int m0 = mt << 4;
  int lane = threadIdx.x & 31;
  int nn = lane & 15, hh = lane >> 4;

  v16h qa0 = frag_rowmajor(Qb, m0, ld, 0);
  v16h qa1 = frag_rowmajor(Qb, m0, ld, 32);

  float rowmax[8], rowsum[8];
  v8f o[4] = {};
#pragma unroll
  for (int r = 0; r < 8; ++r) { rowmax[r] = -INFINITY; rowsum[r] = 0.0f; }

  int nchunks = ((m0 + 15) >> 5) + 1;
  const float scale = 0.125f;  // 1/sqrt(64)
  for (int c = 0; c < nchunks; ++c) {
    int kb = c << 5;
    v8f s0 = {}, s1 = {};
    s0 = wmma16(qa0, frag_rowmajor(Kb, kb, ld, 0), s0);
    s0 = wmma16(qa1, frag_rowmajor(Kb, kb, ld, 32), s0);
    s1 = wmma16(qa0, frag_rowmajor(Kb, kb + 16, ld, 0), s1);
    s1 = wmma16(qa1, frag_rowmajor(Kb, kb + 16, ld, 32), s1);
#pragma unroll
    for (int r = 0; r < 8; ++r) {
      int qi = m0 + r + 8 * hh;
      float v0 = s0[r] * scale; if (kb + nn > qi) v0 = -INFINITY;
      float v1 = s1[r] * scale; if (kb + 16 + nn > qi) v1 = -INFINITY;
      float mx = fmaxf(v0, v1);
#pragma unroll
      for (int w = 1; w < 16; w <<= 1) mx = fmaxf(mx, __shfl_xor(mx, w, 16));
      float nm = fmaxf(rowmax[r], mx);
      float corr = expf(rowmax[r] - nm);
      rowmax[r] = nm;
      float e0 = expf(v0 - nm), e1 = expf(v1 - nm);
      float ps = e0 + e1;
#pragma unroll
      for (int w = 1; w < 16; w <<= 1) ps += __shfl_xor(ps, w, 16);
      rowsum[r] = rowsum[r] * corr + ps;
#pragma unroll
      for (int j = 0; j < 4; ++j) o[j][r] *= corr;
      pbuf[wid][(r + 8 * hh) * 32 + nn] = (_Float16)e0;
      pbuf[wid][(r + 8 * hh) * 32 + 16 + nn] = (_Float16)e1;
    }
    __asm__ volatile("s_wait_dscnt 0" ::: "memory");  // same-wave LDS turnaround
    v16h pa = frag_rowmajor(&pbuf[wid][0], 0, 32, 0);
#pragma unroll
    for (int j = 0; j < 4; ++j)
      o[j] = wmma16(pa, frag_gather(Vb, kb, ld, j * 16), o[j]);
  }
#pragma unroll
  for (int j = 0; j < 4; ++j) {
#pragma unroll
    for (int r = 0; r < 8; ++r) {
      int row = m0 + r + 8 * hh;
      float v = o[j][r] / rowsum[r];
      aoh[((size_t)b * kT + row) * kD + h * kDH + j * 16 + nn] = (_Float16)v;
    }
  }
}

// out = LayerNorm(X + R) row-wise over 256; one wave per row.
__global__ void add_ln_kernel(const float* __restrict__ X, const float* __restrict__ R,
                              const float* __restrict__ g, const float* __restrict__ bt,
                              float* __restrict__ outf, _Float16* __restrict__ outh, int M) {
  int wid = (blockIdx.x * blockDim.x + threadIdx.x) >> 5;
  int lane = threadIdx.x & 31;
  if (wid >= M) return;
  const float* xr = X + (size_t)wid * kD;
  const float* rr = R + (size_t)wid * kD;
  float vals[8], sum = 0.0f, sq = 0.0f;
#pragma unroll
  for (int j = 0; j < 8; ++j) {
    float v = xr[lane + j * 32] + rr[lane + j * 32];
    vals[j] = v; sum += v; sq += v * v;
  }
#pragma unroll
  for (int w = 1; w < 32; w <<= 1) { sum += __shfl_xor(sum, w, 32); sq += __shfl_xor(sq, w, 32); }
  float mean = sum * (1.0f / 256.0f);
  float var  = sq * (1.0f / 256.0f) - mean * mean;
  float rstd = rsqrtf(var + 1e-5f);
#pragma unroll
  for (int j = 0; j < 8; ++j) {
    int c = lane + j * 32;
    float v = (vals[j] - mean) * rstd * g[c] + bt[c];
    if (outf) outf[(size_t)wid * kD + c] = v;
    if (outh) outh[(size_t)wid * kD + c] = (_Float16)v;
  }
}

// xgh = f16(x1 + 0.3 * guid[b])
__global__ void guid_add_kernel(const float* __restrict__ x1, const float* __restrict__ guid,
                                _Float16* __restrict__ xgh, long n) {
  long i = (long)blockIdx.x * blockDim.x + threadIdx.x;
  if (i >= n) return;
  int b = (int)(i >> 17);   // T*D = 131072
  int c = (int)(i & 255);
  xgh[i] = (_Float16)(x1[i] + 0.3f * guid[b * kD + c]);
}

// -------------------------- sequential scan: one persistent block, 512 threads
__global__ void scan_kernel(
    const _Float16* __restrict__ x2h, const float* __restrict__ x2f,
    const float* __restrict__ pg,
    const _Float16* __restrict__ Wq, const float* __restrict__ bq,
    const _Float16* __restrict__ Wk, const float* __restrict__ bk,
    const _Float16* __restrict__ Wv, const float* __restrict__ bv,
    const _Float16* __restrict__ Wo, const float* __restrict__ bo,
    const _Float16* __restrict__ Wt1, const float* __restrict__ bt1,
    const _Float16* __restrict__ Wt2, const float* __restrict__ bt2,
    const float* __restrict__ ln3g, const float* __restrict__ ln3b,
    const float* __restrict__ ln4g, const float* __restrict__ ln4b,
    const float* __restrict__ h0, const float* __restrict__ summ0,
    float* __restrict__ kring, float* __restrict__ vring,
    float* __restrict__ outFinal, float* __restrict__ outSumm) {
  (void)x2f;
  extern __shared__ __align__(16) char smem[];
  float* sh_h    = (float*)smem;            // [32][256]
  float* sh_summ = sh_h + 8192;             // [32][256]
  float* sh_a    = sh_summ + 8192;          // scratch [32][256]
  float* sh_b    = sh_a + 8192;             // scratch [32][256]
  _Float16* act_a = (_Float16*)(sh_b + 8192);  // [32][512]
  _Float16* act_b = act_a + 16384;             // [32][512]
  _Float16* act_x = act_b + 16384;             // double-buffered x_t tile [2][32][256]

  int tid = threadIdx.x;
  int wid = tid >> 5, lane = tid & 31;
  int nn = lane & 15, hh = lane >> 4;

  if (wid == 0) {
    __builtin_prefetch(Wt1, 0, 3);
    __builtin_prefetch(Wt2, 0, 3);
    __builtin_prefetch(Wq, 0, 3);
    __builtin_prefetch(Wk, 0, 3);
  }

  for (int i = tid; i < 8192; i += 512) { sh_h[i] = h0[i]; sh_summ[i] = summ0[i]; }
  for (int i = tid; i < 9 * 8192; i += 512) { kring[i] = 0.0f; vring[i] = 0.0f; }

#ifdef SCAN_TDM
  // DMA the first x_t tile (32 rows of 256 f16, row stride T*D) while we init
  if (wid == 0)
    tdm_load_tile_f16((unsigned)(size_t)act_x, x2h, kD, kB, (unsigned long)kT * kD);
#endif
  __syncthreads();

  for (int t = 0; t < kT; ++t) {
    const _Float16* xt = act_x + (t & 1) * 8192;
#ifdef SCAN_TDM
    // prefetch next tile via TDM; ensure current tile is complete
    if (wid == 0) {
      if (t + 1 < kT) {
        tdm_load_tile_f16((unsigned)(size_t)(act_x + ((t + 1) & 1) * 8192),
                          x2h + (size_t)(t + 1) * kD, kD, kB, (unsigned long)kT * kD);
        wait_tensorcnt1();
      } else {
        wait_tensorcnt0();
      }
    }
#else
    for (int i = tid; i < 8192; i += 512)
      act_x[(t & 1) * 8192 + i] = x2h[(size_t)(i >> 8) * kT * kD + (size_t)t * kD + (i & 255)];
#endif
    // 1: stage summ (f16) as GEMM A operand
    for (int i = tid; i < 8192; i += 512)
      act_a[(i >> 8) * 512 + (i & 255)] = (_Float16)sh_summ[i];
    __syncthreads();

    // 2: qf = fmap(x_t@Wq+bq) -> sh_a ; summary key/value -> ring slot 8
    for (int tix = wid; tix < 96; tix += 16) {
      int which = tix / 32, tt = tix % 32;
      int m0 = (tt & 1) << 4, n0 = (tt >> 1) << 4;
      v8f acc;
      if (which == 0)      acc = tile_mm(xt, kD, m0, Wq, kD, n0, kD);
      else if (which == 1) acc = tile_mm(act_a, 512, m0, Wk, kD, n0, kD);
      else                 acc = tile_mm(act_a, 512, m0, Wv, kD, n0, kD);
      const float* bias = (which == 0) ? bq : ((which == 1) ? bk : bv);
#pragma unroll
      for (int r = 0; r < 8; ++r) {
        int row = m0 + r + 8 * hh, col = n0 + nn;
        float v = acc[r] + bias[col];
        if (which == 0)      sh_a[row * kD + col] = fmapf(v);
        else if (which == 1) kring[8 * 8192 + row * kD + col] = fmapf(v);
        else                 vring[8 * 8192 + row * kD + col] = v;
      }
    }
    __threadfence_block();
    __syncthreads();

    // 3: linear attention  num = sum_s (q.k_s) v_s ; 4 lanes per (b,head)
    {
      int pair = tid >> 2, sub = tid & 3;
      int b = pair >> 2, hd = pair & 3;
      int base = b * kD + hd * kDH;
      float w[9];
#pragma unroll
      for (int s = 0; s < 9; ++s) {
        float p = 0.0f;
        for (int d = sub * 16; d < sub * 16 + 16; ++d)
          p += sh_a[base + d] * kring[s * 8192 + base + d];
        p += __shfl_xor(p, 1, 4);
        p += __shfl_xor(p, 2, 4);
        w[s] = p;
      }
      float den = 0.0f;
#pragma unroll
      for (int s = 0; s < 9; ++s) den += w[s];
      float inv = 1.0f / (den + 1e-6f);
      for (int e = sub * 16; e < sub * 16 + 16; ++e) {
        float num = 0.0f;
#pragma unroll
        for (int s = 0; s < 9; ++s) num += w[s] * vring[s * 8192 + base + e];
        act_a[b * 512 + hd * kDH + e] = (_Float16)(num * inv);
      }
    }
    __syncthreads();

    // 4: ctx = attnout @ Wo^T + bo -> sh_b
    for (int tix = wid; tix < 32; tix += 16) {
      int m0 = (tix & 1) << 4, n0 = (tix >> 1) << 4;
      v8f acc = tile_mm(act_a, 512, m0, Wo, kD, n0, kD);
#pragma unroll
      for (int r = 0; r < 8; ++r)
        sh_b[(m0 + r + 8 * hh) * kD + n0 + nn] = acc[r] + bo[n0 + nn];
    }
    __syncthreads();

    // 5: summary decay (uses pre-update h)
    if (t > 0)
      for (int i = tid; i < 8192; i += 512)
        sh_summ[i] = 0.9f * sh_summ[i] + 0.1f * sh_h[i];
    __syncthreads();

    // 6: comb = [x_t + 0.3*pg_t , traj] -> act_a [32][512] f16
    for (int i = tid; i < 8192; i += 512) {
      int r = i >> 8, c = i & 255;
      size_t gi = ((size_t)r * kT + t) * kD + c;
      act_a[r * 512 + c] = (_Float16)((float)xt[r * kD + c] + 0.3f * pg[gi]);
      act_a[r * 512 + 256 + c] = (_Float16)((t > 0) ? sh_b[i] : 0.0f);
    }
    __syncthreads();

    // 7: g1 = gelu(comb @ Wt1^T + bt1) -> act_b [32][512] f16
    for (int tix = wid; tix < 64; tix += 16) {
      int m0 = (tix & 1) << 4, n0 = (tix >> 1) << 4;
      v8f acc = tile_mm(act_a, 512, m0, Wt1, 512, n0, 512);
#pragma unroll
      for (int r = 0; r < 8; ++r)
        act_b[(m0 + r + 8 * hh) * 512 + n0 + nn] = (_Float16)geluf(acc[r] + bt1[n0 + nn]);
    }
    __syncthreads();

    // 8: delta = g1 @ Wt2^T + bt2 -> sh_a
    for (int tix = wid; tix < 32; tix += 16) {
      int m0 = (tix & 1) << 4, n0 = (tix >> 1) << 4;
      v8f acc = tile_mm(act_b, 512, m0, Wt2, 512, n0, 512);
#pragma unroll
      for (int r = 0; r < 8; ++r)
        sh_a[(m0 + r + 8 * hh) * kD + n0 + nn] = acc[r] + bt2[n0 + nn];
    }
    __syncthreads();

    // 9: h = LN3(h + 0.5*delta) ; 16 lanes per row
    {
      int row = tid >> 4, cs = tid & 15;
      float sum = 0.0f, sq = 0.0f;
#pragma unroll
      for (int j = 0; j < 16; ++j) {
        int c = cs + j * 16;
        float y = sh_h[row * kD + c] + 0.5f * sh_a[row * kD + c];
        sum += y; sq += y * y;
      }
#pragma unroll
      for (int w = 1; w < 16; w <<= 1) { sum += __shfl_xor(sum, w, 16); sq += __shfl_xor(sq, w, 16); }
      float mean = sum * (1.0f / 256.0f);
      float rstd = rsqrtf(sq * (1.0f / 256.0f) - mean * mean + 1e-5f);
#pragma unroll
      for (int j = 0; j < 16; ++j) {
        int c = cs + j * 16;
        float y = sh_h[row * kD + c] + 0.5f * sh_a[row * kD + c];
        sh_h[row * kD + c] = (y - mean) * rstd * ln3g[c] + ln3b[c];
      }
    }
    __syncthreads();

    // 10: stage new h (f16); 11: its key/value features into ring slot t%W
    for (int i = tid; i < 8192; i += 512)
      act_a[(i >> 8) * 512 + (i & 255)] = (_Float16)sh_h[i];
    __syncthreads();
    int slot = t & (kW - 1);
    for (int tix = wid; tix < 64; tix += 16) {
      int which = tix / 32, tt = tix % 32;
      int m0 = (tt & 1) << 4, n0 = (tt >> 1) << 4;
      v8f acc = tile_mm(act_a, 512, m0, (which == 0) ? Wk : Wv, kD, n0, kD);
      const float* bias = (which == 0) ? bk : bv;
#pragma unroll
      for (int r = 0; r < 8; ++r) {
        int row = m0 + r + 8 * hh, col = n0 + nn;
        float v = acc[r] + bias[col];
        if (which == 0) kring[slot * 8192 + row * kD + col] = fmapf(v);
        else            vring[slot * 8192 + row * kD + col] = v;
      }
    }
    __threadfence_block();
    __syncthreads();
  }

  // final = LN4(h) ; summ out
  {
    int row = tid >> 4, cs = tid & 15;
    float sum = 0.0f, sq = 0.0f;
#pragma unroll
    for (int j = 0; j < 16; ++j) {
      int c = cs + j * 16;
      float y = sh_h[row * kD + c];
      sum += y; sq += y * y;
    }
#pragma unroll
    for (int w = 1; w < 16; w <<= 1) { sum += __shfl_xor(sum, w, 16); sq += __shfl_xor(sq, w, 16); }
    float mean = sum * (1.0f / 256.0f);
    float rstd = rsqrtf(sq * (1.0f / 256.0f) - mean * mean + 1e-5f);
#pragma unroll
    for (int j = 0; j < 16; ++j) {
      int c = cs + j * 16;
      float y = sh_h[row * kD + c];
      outFinal[row * kD + c] = (y - mean) * rstd * ln4g[c] + ln4b[c];
    }
  }
  for (int i = tid; i < 8192; i += 512) outSumm[i] = sh_summ[i];
}

// ---------------------------------------------------------------- launcher

extern "C" void kernel_launch(void* const* d_in, const int* in_sizes, int n_in,
                              void* d_out, int out_size, void* d_ws, size_t ws_size,
                              hipStream_t stream) {
  (void)in_sizes; (void)n_in; (void)out_size; (void)ws_size;
  const float* x_in   = (const float*)d_in[0];
  const float* seq0   = (const float*)d_in[1];
  const float* traj0  = (const float*)d_in[2];
  const float* mha_w  = (const float*)d_in[3];
  const float* mha_b  = (const float*)d_in[4];
  const float* mha_ow = (const float*)d_in[5];
  const float* mha_ob = (const float*)d_in[6];
  const float* ffn_w1 = (const float*)d_in[7];
  const float* ffn_b1 = (const float*)d_in[8];
  const float* ffn_w2 = (const float*)d_in[9];
  const float* ffn_b2 = (const float*)d_in[10];
  const float* ta_qw = (const float*)d_in[11]; const float* ta_qb = (const float*)d_in[12];
  const float* ta_kw = (const float*)d_in[13]; const float* ta_kb = (const float*)d_in[14];
  const float* ta_vw = (const float*)d_in[15]; const float* ta_vb = (const float*)d_in[16];
  const float* ta_ow = (const float*)d_in[17]; const float* ta_ob = (const float*)d_in[18];
  const float* tr_w1 = (const float*)d_in[19]; const float* tr_b1 = (const float*)d_in[20];
  const float* tr_w2 = (const float*)d_in[21]; const float* tr_b2 = (const float*)d_in[22];
  const float* p2s_w = (const float*)d_in[23]; const float* p2s_b = (const float*)d_in[24];
  const float* s2p_w = (const float*)d_in[25]; const float* s2p_b = (const float*)d_in[26];
  const float* ln1_g = (const float*)d_in[27]; const float* ln1_b = (const float*)d_in[28];
  const float* ln2_g = (const float*)d_in[29]; const float* ln2_b = (const float*)d_in[30];
  const float* ln3_g = (const float*)d_in[31]; const float* ln3_b = (const float*)d_in[32];
  const float* ln4_g = (const float*)d_in[33]; const float* ln4_b = (const float*)d_in[34];

  float* out = (float*)d_out;
  float* outFinal = out + kBTD;
  float* outSumm  = outFinal + kBD;

  size_t off = 0;
  auto alloc = [&](size_t bytes) -> void* {
    void* p = (char*)d_ws + off;
    off = (off + bytes + 255) & ~(size_t)255;
    return p;
  };
  auto allocH = [&](long n) { return (_Float16*)alloc((size_t)n * sizeof(_Float16)); };
  auto allocF = [&](long n) { return (float*)alloc((size_t)n * sizeof(float)); };

  _Float16* w_mha = allocH(768 * 256);
  _Float16* w_mow = allocH(256 * 256);
  _Float16* w_f1  = allocH(1024 * 256);
  _Float16* w_f2  = allocH(256 * 1024);
  _Float16* w_taq = allocH(256 * 256);
  _Float16* w_tak = allocH(256 * 256);
  _Float16* w_tav = allocH(256 * 256);
  _Float16* w_tao = allocH(256 * 256);
  _Float16* w_tr1 = allocH(512 * 512);
  _Float16* w_tr2 = allocH(256 * 512);
  _Float16* w_p2s = allocH(256 * 256);
  _Float16* w_s2p = allocH(256 * 256);
  _Float16* xh    = allocH(kBTD);
  _Float16* seqh  = allocH(kBD);
  _Float16* qkvh  = allocH(3 * kBTD);
  _Float16* aoh   = allocH(kBTD);
  float*    ao32  = allocF(kBTD);
  float*    x1f   = allocF(kBTD);
  float*    guid32= allocF(kBD);
  _Float16* xgh   = allocH(kBTD);
  _Float16* h1h   = allocH(4 * kBTD);
  float*    ffn32 = allocF(kBTD);
  _Float16* x2h   = allocH(kBTD);
  float*    pg32  = allocF(kBTD);
  float*    kring = allocF(9 * kBD);
  float*    vring = allocF(9 * kBD);

  auto cast = [&](const float* s, _Float16* d, long n) {
    cast_f16_kernel<<<(unsigned)((n + 255) / 256), 256, 0, stream>>>(s, d, (int)n);
  };
  cast(mha_w, w_mha, 768 * 256);
  cast(mha_ow, w_mow, 256 * 256);
  cast(ffn_w1, w_f1, 1024 * 256);
  cast(ffn_w2, w_f2, 256 * 1024);
  cast(ta_qw, w_taq, 256 * 256);
  cast(ta_kw, w_tak, 256 * 256);
  cast(ta_vw, w_tav, 256 * 256);
  cast(ta_ow, w_tao, 256 * 256);
  cast(tr_w1, w_tr1, 512 * 512);
  cast(tr_w2, w_tr2, 256 * 512);
  cast(p2s_w, w_p2s, 256 * 256);
  cast(s2p_w, w_s2p, 256 * 256);
  cast(x_in, xh, kBTD);
  cast(seq0, seqh, kBD);

  auto gemm = [&](const _Float16* A, int lda, const _Float16* Wm, int ldw,
                  const float* bias, float* Cf, _Float16* Ch,
                  int M, int N, int K, int act) {
    int waves = (M >> 4) * (N >> 6);
    gemm_nt_kernel<<<(waves + 3) / 4, 128, 0, stream>>>(A, lda, Wm, ldw, bias, Cf, Ch, M, N, K, act);
  };

  // ---- parallel phase ----
  gemm(xh, 256, w_mha, 256, mha_b, nullptr, qkvh, kBT, 768, 256, 0);          // qkv
  flash_attn_kernel<<<kB * kH * 32 / 4, 128, 0, stream>>>(qkvh, aoh);         // attention
  gemm(aoh, 256, w_mow, 256, mha_ob, ao32, nullptr, kBT, 256, 256, 0);        // out proj
  add_ln_kernel<<<kBT / 8, 256, 0, stream>>>(x_in, ao32, ln1_g, ln1_b, x1f, nullptr, kBT);
  gemm(seqh, 256, w_s2p, 256, s2p_b, guid32, nullptr, kB, 256, 256, 0);       // guidance
  guid_add_kernel<<<(unsigned)((kBTD + 255) / 256), 256, 0, stream>>>(x1f, guid32, xgh, kBTD);
  gemm(xgh, 256, w_f1, 256, ffn_b1, nullptr, h1h, kBT, 1024, 256, 1);         // ffn up + gelu
  gemm(h1h, 1024, w_f2, 1024, ffn_b2, ffn32, nullptr, kBT, 256, 1024, 0);     // ffn down
  add_ln_kernel<<<kBT / 8, 256, 0, stream>>>(x1f, ffn32, ln2_g, ln2_b, out, x2h, kBT); // x (output 1)
  gemm(x2h, 256, w_p2s, 256, p2s_b, pg32, nullptr, kBT, 256, 256, 0);         // parallel->seq guide

  // ---- sequential phase (persistent block) ----
  size_t smem = 4 * 8192 * sizeof(float) + 2 * 16384 * sizeof(_Float16)
              + 2 * 8192 * sizeof(_Float16);                                  // 224 KB
  scan_kernel<<<1, 512, smem, stream>>>(
      x2h, out, pg32,
      w_taq, ta_qb, w_tak, ta_kb, w_tav, ta_vb, w_tao, ta_ob,
      w_tr1, tr_b1, w_tr2, tr_b2,
      ln3_g, ln3_b, ln4_g, ln4_b,
      seq0, traj0, kring, vring, outFinal, outSumm);
}
